// LocallyHierarchicalNet_6820408066061
// MI455X (gfx1250) — matrix operand order repaired and checked by reading, not compiled, not run on hardware
//
#include <hip/hip_runtime.h>
#include <hip/hip_bf16.h>

// MI455X (gfx1250) implementation.
// Requires ws_size >= ~52 MiB (activation ping-pong in bf16 + fp32 head input).
//
// Layout conventions:
//   Activations between layers are stored bf16 as (B=128, Dd_next, K=512) with
//   K = (channel, sub-pos) contiguous -> exactly the A-matrix K ordering, so
//   WMMA A fragments are contiguous 16B loads.
//   Weights stay fp32 in HBM (streamed once, non-temporal, converted to bf16
//   in registers feeding v_wmma_f32_16x16x32_bf16).
//
// Scheduling:
//   DD >= 32 : MSPLIT=1, one WG per position d -> each weight byte fetched
//              from HBM exactly once (HBM-bound phase, ~268MB @ 23.3TB/s).
//   DD <= 16 : MSPLIT=4, WG owns 32 batch rows -> 4x parallelism, 4x less
//              serial work per wave (latency-bound phase; weights L2-resident).

typedef __attribute__((ext_vector_type(16))) __bf16 bf16x16;
typedef __attribute__((ext_vector_type(8)))  __bf16 bf16x8;
typedef __attribute__((ext_vector_type(8)))  float  f32x8;
typedef __attribute__((ext_vector_type(2)))  float  f32x2;

// ---------------------------------------------------------------------------
// Layer 0: (B,3,1024) x fp32, w0 (256,3,512,2) fp32 -> bf16 act (B,256,512)
// K is only 6, so plain VALU FMAs; each block handles one d1 (pair of output
// positions) so the two bf16 outputs pack into a single dword store.
// ---------------------------------------------------------------------------
__global__ __launch_bounds__(256) void lhn_layer0(const float* __restrict__ x,
                                                  const float* __restrict__ w0,
                                                  __bf16* __restrict__ out) {
  const int d1 = blockIdx.x;   // 0..255  (position in layer-1 input)
  const int o  = threadIdx.x;  // 0..255  (output channel)

  float4 wr[3];
#pragma unroll
  for (int c = 0; c < 3; ++c)
    wr[c] = *reinterpret_cast<const float4*>(w0 + (size_t)(o * 3 + c) * 1024 + 4 * d1);

  const float s = 0.57735026918962576f;  // 1/sqrt(3)
  for (int b = 0; b < 128; ++b) {
    float e = 0.f, od = 0.f;
#pragma unroll
    for (int c = 0; c < 3; ++c) {
      float4 xv = *reinterpret_cast<const float4*>(x + (size_t)(b * 3 + c) * 1024 + 4 * d1);
      e  += xv.x * wr[c].x + xv.y * wr[c].y;   // d0 = 2*d1   (kk=0,1)
      od += xv.z * wr[c].z + xv.w * wr[c].w;   // d0 = 2*d1+1 (kk=0,1)
    }
    e *= s; od *= s;
    e  = e  > 0.f ? e  : 0.f;
    od = od > 0.f ? od : 0.f;
    __bf16 p0 = (__bf16)e, p1 = (__bf16)od;
    unsigned u = (unsigned)__builtin_bit_cast(unsigned short, p0) |
                 ((unsigned)__builtin_bit_cast(unsigned short, p1) << 16);
    *reinterpret_cast<unsigned*>(out + (size_t)(b * 256 + d1) * 512 + 2 * o) = u;
  }
}

// ---------------------------------------------------------------------------
// Layers 1..9: per position d, GEMM (M=128,B-dim) x (K=512) x (N=256,out-ch)
// Wave w: N-tiles {2w, 2w+1}; M-tiles of this WG's slice. DD is a template
// parameter so all weight loads become base+immediate-offset (max 255*DD*8 =
// 510KB < 24-bit ioffset) with zero per-load address VALU.
// ---------------------------------------------------------------------------
template <int DD, int MSPLIT, bool LAST>
__global__ __launch_bounds__(256) void lhn_layer(const __bf16* __restrict__ act_in,
                                                 const float* __restrict__ w,
                                                 __bf16* __restrict__ act_out,
                                                 float* __restrict__ y_out) {
  constexpr int MT   = 8 / MSPLIT;   // M-tiles per workgroup
  constexpr int ROWS = MT * 16;      // batch rows per workgroup

  const int d   = blockIdx.x;
  const int m0  = blockIdx.y * ROWS;  // first batch row of this WG
  const int tid = threadIdx.x;
  const int wv  = tid >> 5;    // wave 0..7
  const int ln  = tid & 31;
  const int r   = ln & 15;     // N-column / M-row within tile
  const int h   = ln >> 4;     // lane half

  // A staging: ROWS x 128 K-chunk, row stride 136 bf16 (272B -> 4-dword
  // bank skew per row => conflict-free ds_load_b128).
  __shared__ __bf16 ldsA[ROWS * 136];

  // Per-lane weight base pointers for the two N-tiles; everything else is a
  // compile-time immediate offset: channel c contributes c*DD*2 floats.
  const float* wbase[2];
#pragma unroll
  for (int t = 0; t < 2; ++t) {
    const int n = wv * 32 + t * 16 + r;
    wbase[t] = w + ((size_t)n * 256 + h * 8) * (DD * 2) + (size_t)d * 2;
  }

  f32x8 acc[MT][2];
#pragma unroll
  for (int mt = 0; mt < MT; ++mt)
#pragma unroll
    for (int t = 0; t < 2; ++t)
#pragma unroll
      for (int j = 0; j < 8; ++j) acc[mt][t][j] = 0.f;

#pragma unroll
  for (int kc = 0; kc < 4; ++kc) {   // K chunks of 128
    const int ks0 = kc * 128;
    __syncthreads();
    // cooperative stage: ROWS x 128 bf16, 16B per transfer
#pragma unroll
    for (int it = 0; it < MT; ++it) {
      int i   = tid + it * 256;    // 0..ROWS*16-1
      int row = i >> 4;
      int c16 = i & 15;
      const uint4* gp = reinterpret_cast<const uint4*>(
          act_in + ((size_t)(m0 + row) * DD + d) * 512 + ks0 + c16 * 8);
      *reinterpret_cast<uint4*>(&ldsA[row * 136 + c16 * 8]) = *gp;
    }
    __syncthreads();

#pragma unroll
    for (int s = 0; s < 4; ++s) {    // K-steps of 32
      const int ks     = ks0 + s * 32;
      const int klocal = s * 32;

      // B fragments: lane = column n, K = 16 consecutive values per lane half
      // (channels c0..c0+7, two fp32 each, stride DD*2 floats), NT-load + cvt.
      bf16x16 bfrag[2];
#pragma unroll
      for (int t = 0; t < 2; ++t) {
#pragma unroll
        for (int q = 0; q < 8; ++q) {
          const int c = (ks >> 1) + q;         // compile-time immediate
          f32x2 v = __builtin_nontemporal_load(
              reinterpret_cast<const f32x2*>(wbase[t] + (size_t)c * (DD * 2)));
          bfrag[t][2 * q]     = (__bf16)v.x;
          bfrag[t][2 * q + 1] = (__bf16)v.y;
        }
      }

      // A fragments from LDS, software-pipelined one M-tile ahead so the
      // ds_load latency hides under the previous tile's WMMAs.
      bf16x8 alo = *reinterpret_cast<const bf16x8*>(&ldsA[r * 136 + klocal + h * 16]);
      bf16x8 ahi = *reinterpret_cast<const bf16x8*>(&ldsA[r * 136 + klocal + h * 16 + 8]);
#pragma unroll
      for (int mt = 0; mt < MT; ++mt) {
        bf16x16 afrag;
#pragma unroll
        for (int q = 0; q < 8; ++q) { afrag[q] = alo[q]; afrag[q + 8] = ahi[q]; }
        if (mt + 1 < MT) {
          const int row = (mt + 1) * 16 + r;
          alo = *reinterpret_cast<const bf16x8*>(&ldsA[row * 136 + klocal + h * 16]);
          ahi = *reinterpret_cast<const bf16x8*>(&ldsA[row * 136 + klocal + h * 16 + 8]);
        }
        acc[mt][0] = __builtin_amdgcn_wmma_f32_16x16x32_bf16(
            false, afrag, false, bfrag[0], (short)0, acc[mt][0], false, false);
        acc[mt][1] = __builtin_amdgcn_wmma_f32_16x16x32_bf16(
            false, afrag, false, bfrag[1], (short)0, acc[mt][1], false, false);
      }
    }
  }

  // Epilogue: scale 1/sqrt(256), ReLU, scatter to next layer's A layout
  // (or fp32 head input for the last layer). C/D layout: M = j + 8h, N = r.
  const float scale = 0.0625f;
  if (!LAST) {
    const int Dd2 = DD >> 1;
    const int d2  = d >> 1;
    const int kk  = d & 1;
#pragma unroll
    for (int mt = 0; mt < MT; ++mt)
#pragma unroll
      for (int t = 0; t < 2; ++t) {
        const int o = wv * 32 + t * 16 + r;
#pragma unroll
        for (int j = 0; j < 8; ++j) {
          const int b = m0 + mt * 16 + h * 8 + j;
          float v = acc[mt][t][j] * scale;
          v = v > 0.f ? v : 0.f;
          act_out[((size_t)b * Dd2 + d2) * 512 + 2 * o + kk] = (__bf16)v;
        }
      }
  } else {  // DD == 1: mean over 1 position == identity; emit fp32 (B,256)
#pragma unroll
    for (int mt = 0; mt < MT; ++mt)
#pragma unroll
      for (int t = 0; t < 2; ++t) {
        const int o = wv * 32 + t * 16 + r;
#pragma unroll
        for (int j = 0; j < 8; ++j) {
          const int b = m0 + mt * 16 + h * 8 + j;
          float v = acc[mt][t][j] * scale;
          v = v > 0.f ? v : 0.f;
          y_out[(size_t)b * 256 + o] = v;
        }
      }
  }
}

// ---------------------------------------------------------------------------
// Head: (128,256) @ (256,10) / 10 -> (128,10). Tiny; one wave per batch row.
// ---------------------------------------------------------------------------
__global__ __launch_bounds__(32) void lhn_head(const float* __restrict__ y,
                                               const float* __restrict__ beta,
                                               float* __restrict__ out) {
  const int b  = blockIdx.x;
  const int ln = threadIdx.x;
  float acc[10];
#pragma unroll
  for (int o = 0; o < 10; ++o) acc[o] = 0.f;
  for (int c = ln; c < 256; c += 32) {
    float yv = y[(size_t)b * 256 + c];
#pragma unroll
    for (int o = 0; o < 10; ++o) acc[o] += yv * beta[c * 10 + o];
  }
#pragma unroll
  for (int o = 0; o < 10; ++o) {
    float v = acc[o];
#pragma unroll
    for (int off = 16; off > 0; off >>= 1) v += __shfl_down(v, off, 32);
    if (ln == 0) out[b * 10 + o] = v * 0.1f;
  }
}

template <int DD, int MSPLIT, bool LAST>
static inline void launch_layer(const __bf16* in, const float* w,
                                __bf16* outb, float* outf, hipStream_t stream) {
  lhn_layer<DD, MSPLIT, LAST>
      <<<dim3(DD, MSPLIT), dim3(256), 0, stream>>>(in, w, outb, outf);
}

extern "C" void kernel_launch(void* const* d_in, const int* in_sizes, int n_in,
                              void* d_out, int out_size, void* d_ws, size_t ws_size,
                              hipStream_t stream) {
  const float* x = (const float*)d_in[0];
  const float* w[10];
  for (int l = 0; l < 10; ++l) w[l] = (const float*)d_in[1 + l];
  const float* beta = (const float*)d_in[11];
  float* out = (float*)d_out;

  char* ws = (char*)d_ws;
  __bf16* actA = (__bf16*)(ws);                      // 32 MiB: layer-0 output
  __bf16* actB = (__bf16*)(ws + (34ull << 20));      // <=16 MiB
  float*  yv   = (float*)(ws + (51ull << 20));       // 128 KiB: head input

  lhn_layer0<<<dim3(256), dim3(256), 0, stream>>>(x, w[0], actA);

  // ping-pong: odd layers read actA/write actB, even layers the reverse.
  // Big layers: MSPLIT=1 (fetch-once weight streaming, HBM-bound).
  // Tail layers: MSPLIT=4 (latency-bound; weights L2-resident anyway).
  launch_layer<256, 1, false>(actA, w[1], actB, nullptr, stream);
  launch_layer<128, 1, false>(actB, w[2], actA, nullptr, stream);
  launch_layer< 64, 1, false>(actA, w[3], actB, nullptr, stream);
  launch_layer< 32, 1, false>(actB, w[4], actA, nullptr, stream);
  launch_layer< 16, 4, false>(actA, w[5], actB, nullptr, stream);
  launch_layer<  8, 4, false>(actB, w[6], actA, nullptr, stream);
  launch_layer<  4, 4, false>(actA, w[7], actB, nullptr, stream);
  launch_layer<  2, 4, false>(actB, w[8], actA, nullptr, stream);
  launch_layer<  1, 4, true >(actA, w[9], nullptr, yv, stream);

  lhn_head<<<dim3(128), dim3(32), 0, stream>>>(yv, beta, out);
}
